// VPTLSTM_71949292142746
// MI455X (gfx1250) — matrix operand order repaired and checked by reading, not compiled, not run on hardware
//
#include <hip/hip_runtime.h>
#include <hip/hip_bf16.h>
#include <math.h>

// ---------------------------------------------------------------------------
// Social-LSTM on MI455X (gfx1250), wave32 + WMMA f16->f32.
//
// Structure:
//  * pack_kernel (7 launches): f32 weights -> f16, repacked into WMMA-B
//    fragment-linear order: frag[(tile*nChunks+chunk)][lane][16 halves],
//    so each B-fragment load is lane-contiguous 32B (coalesced 1KB/wave).
//  * social_lstm_kernel: 16 blocks x 128 threads (4 waves). Block b owns
//    vehicles [16b,16b+16) for all 99 steps (vehicle recurrences are
//    independent). All activations live in ~140KB dynamic LDS.
//
// conv1 factorization:  t1[v,o,y,x] = relu(b1[o] + sum_p occ[v,pos] * A1[v,p,o])
//   where A1 = h @ W1r^T  (GEMM, N = p*64+o, K = 128)   -> WMMA
// conv2 as im2col GEMM over the contiguous 960-element window of the
//   LDS t1 layout (v, y, x, o).
// ---------------------------------------------------------------------------

typedef _Float16 h16;
typedef __attribute__((ext_vector_type(16))) _Float16 v16h;
typedef __attribute__((ext_vector_type(8)))  _Float16 v8h;
typedef __attribute__((ext_vector_type(8)))  float    v8f;

// ---- WMMA fragment helpers (contiguous k = 16*(L/16) + s in A and B, wave32) ----------

// A-matrix 16x32 f16: lane L holds row m=L%16; K range [16*kh,16*kh+16)
// with kh=L/16, packed as K-pairs into 8 VGPRs.
__device__ __forceinline__ v16h load_a_frag(const h16* rowBase, int strideH, int lane) {
    int m  = lane & 15;
    int kh = lane >> 4;
    const h16* p = rowBase + m * strideH + kh * 16;
    union { v16h v; v8h h[2]; } u;
    u.h[0] = *(const v8h*)(p);        // K = 16kh .. 16kh+7
    u.h[1] = *(const v8h*)(p + 8);    // K = 16kh+8 .. 16kh+15
    return u.v;
}

// B-matrix 32x16 f16 from fragment-packed weights: lane L = 32 contiguous B.
__device__ __forceinline__ v16h load_b_frag(const v16h* __restrict__ packed, int frag, int lane) {
    return packed[frag * 32 + lane];
}

__device__ __forceinline__ v8f wmma_f16(v16h a, v16h b, v8f c) {
    // (neg_a, A, neg_b, B, c_mod, C, reuse_a, reuse_b)
    return __builtin_amdgcn_wmma_f32_16x16x32_f16(false, a, false, b, (short)0, c, false, false);
}

__device__ __forceinline__ v8f fill8(float x) {
    v8f a;
#pragma unroll
    for (int j = 0; j < 8; ++j) a[j] = x;
    return a;
}

__device__ __forceinline__ float sigmoidf_(float x) { return 1.f / (1.f + __expf(-x)); }

// ---- weight pre-packing ----------------------------------------------------
// dst element idx -> (frag, lane, j); logical (n, k) = (tile*16 + lane%16,
// chunk*32 + (lane/16)*16 + j). mode selects the source-weight gather.
__global__ void pack_kernel(const float* __restrict__ src, h16* __restrict__ dst,
                            int mode, int chunks, int total) {
    int idx = blockIdx.x * blockDim.x + threadIdx.x;
    if (idx >= total) return;
    int frag = idx >> 9, r = idx & 511;
    int lane = r >> 4, j = r & 15;
    int tn = frag / chunks, kc = frag % chunks;
    int n = tn * 16 + (lane & 15);
    int k = kc * 32 + (lane >> 4) * 16 + j;
    float val = 0.f;
    switch (mode) {
        case 0:  // W_emb (64,9), K padded 9->32
            if (k < 9) val = src[n * 9 + k];
            break;
        case 1: {  // conv1_w (64,128,5,3) -> W1r[n = p*64+o][k=c]
            int p = n >> 6, o = n & 63, ky = p / 3, kx = p % 3;
            val = src[((o * 128 + k) * 5 + ky) * 3 + kx];
        } break;
        case 2: {  // conv2_w (32,64,5,3) -> W2r[n = o2][k = ky*192+kx*64+o1]
            int ky = k / 192, rem = k % 192, kx = rem / 64, o1 = rem & 63;
            val = src[((n * 64 + o1) * 5 + ky) * 3 + kx];
        } break;
        case 3:  // plain (N,128) row-major (W_soc, W_ih, W_hh)
            val = src[n * 128 + k];
            break;
        case 4:  // W_out (5,128), N padded 5->16
            if (n < 5) val = src[n * 128 + k];
            break;
    }
    dst[idx] = (h16)val;
}

// ---- main persistent kernel ------------------------------------------------
__global__ __launch_bounds__(128, 1)
void social_lstm_kernel(const float* __restrict__ x_seq, const int* __restrict__ grids,
                        const float* __restrict__ h0,    const float* __restrict__ c0,
                        const float* __restrict__ b_emb, const float* __restrict__ b1g,
                        const float* __restrict__ b2g,   const float* __restrict__ b_soc,
                        const float* __restrict__ b_ih,  const float* __restrict__ b_hh,
                        const float* __restrict__ b_out,
                        const v16h* __restrict__ Wemb_p, const v16h* __restrict__ W1_p,
                        const v16h* __restrict__ W2_p,   const v16h* __restrict__ Wsoc_p,
                        const v16h* __restrict__ Wih_p,  const v16h* __restrict__ Whh_p,
                        const v16h* __restrict__ Wout_p,
                        float* __restrict__ out)
{
    extern __shared__ char smem[];
    h16*   sA1  = (h16*)smem;            // 16 x 960  (v, p*64+o)      30720 B
    h16*   sT1  = sA1 + 16 * 960;        // 16 x 1536 (v, y, x, o)     49152 B
    h16*   sT2  = sT1 + 16 * 1536;       // 16 x 128  (v, o2*4+y2)      4096 B
    h16*   sX   = sT2 + 16 * 128;        // 16 x 128  [inp_emb|soc]     4096 B
    h16*   sXin = sX  + 16 * 128;        // 16 x 32   frame padded      1024 B
    h16*   sH   = sXin + 16 * 32;        // 16 x 128  hidden (f16)      4096 B
    float* sG   = (float*)(sH + 16*128); // 16 x 512  gates            32768 B
    float* sC   = sG + 16 * 512;         // 16 x 128  cell              8192 B
    float* sOcc = sC + 16 * 128;         // 16 x 96   occupancy         6144 B
    float* sB   = sOcc + 16 * 96;        // 768 floats of biases        3072 B
    float* sB1   = sB;                   // 64
    float* sB2   = sB + 64;              // 32
    float* sBemb = sB + 96;              // 64
    float* sBsoc = sB + 160;             // 64
    float* sBg   = sB + 224;             // 512 (b_ih + b_hh)
    float* sBout = sB + 736;             // 16 (padded)

    const int tid  = threadIdx.x;
    const int lane = tid & 31;
    const int wave = tid >> 5;
    const int v0   = blockIdx.x * 16;

    for (int i = tid; i < 64; i += 128) { sB1[i] = b1g[i]; sBemb[i] = b_emb[i]; sBsoc[i] = b_soc[i]; }
    for (int i = tid; i < 32; i += 128) sB2[i] = b2g[i];
    for (int i = tid; i < 512; i += 128) sBg[i] = b_ih[i] + b_hh[i];
    for (int i = tid; i < 16; i += 128) sBout[i] = (i < 5) ? b_out[i] : 0.f;
    for (int i = tid; i < 2048; i += 128) {
        int v = i >> 7, r = i & 127;
        sH[i] = (h16)h0[(v0 + v) * 128 + r];
        sC[i] = c0[(v0 + v) * 128 + r];
    }
    __syncthreads();

    for (int t = 0; t < 99; ++t) {
        // -------- stage frame + occupancy into LDS --------
        for (int i = tid; i < 512; i += 128) {
            int v = i >> 5, k = i & 31;
            sXin[i] = (h16)((k < 9) ? x_seq[(t * 256 + v0 + v) * 9 + k] : 0.f);
        }
        for (int i = tid; i < 16 * 95; i += 128) {
            int v = i / 95, g = i % 95;
            sOcc[v * 96 + g] = (grids[(t * 256 + v0 + v) * 95 + g] != -1) ? 1.f : 0.f;
        }
        if (t + 1 < 99) {  // pull next step's inputs toward the caches
            __builtin_prefetch(&x_seq[((t + 1) * 256 + v0 + (tid & 15)) * 9], 0, 1);
            __builtin_prefetch(&grids[((t + 1) * 256 + v0 + (tid & 15)) * 95], 0, 1);
        }
        __syncthreads();

        // -------- inp_emb = relu(frame @ W_emb^T + b_emb): 4 tiles, 1 chunk --------
        {
            int tn = wave;
            v8f acc = fill8(sBemb[tn * 16 + (lane & 15)]);
            acc = wmma_f16(load_a_frag(sXin, 32, lane), load_b_frag(Wemb_p, tn, lane), acc);
            int n = lane & 15, mh = lane >> 4;
#pragma unroll
            for (int j = 0; j < 8; ++j)
                sX[(j + mh * 8) * 128 + tn * 16 + n] = (h16)fmaxf(acc[j], 0.f);
        }
        // -------- A1 = h @ W1r^T : 60 tiles x 4 K-chunks --------
        for (int tn = wave; tn < 60; tn += 4) {
            v8f acc = fill8(0.f);
#pragma unroll
            for (int kc = 0; kc < 4; ++kc)
                acc = wmma_f16(load_a_frag(sH + kc * 32, 128, lane),
                               load_b_frag(W1_p, tn * 4 + kc, lane), acc);
            int n = lane & 15, mh = lane >> 4;
#pragma unroll
            for (int j = 0; j < 8; ++j)
                sA1[(j + mh * 8) * 960 + tn * 16 + n] = (h16)acc[j];
        }
        __syncthreads();

        // -------- occupancy-weighted 15-tap reduction: t1 = relu(b1 + occ*A1) ------
        for (int cidx = tid; cidx < 384; cidx += 128) {
            int v = cidx / 24, yx = cidx % 24, y = yx / 3, x = yx % 3;
            float occ[15];
#pragma unroll
            for (int p = 0; p < 15; ++p)
                occ[p] = sOcc[v * 96 + (2 * y + p / 3) * 5 + (x + p % 3)];
            const h16* arow = sA1 + v * 960;
            h16* trow = sT1 + v * 1536 + yx * 64;
#pragma unroll
            for (int ob = 0; ob < 4; ++ob) {
                float s[16];
#pragma unroll
                for (int i = 0; i < 16; ++i) s[i] = sB1[ob * 16 + i];
                for (int p = 0; p < 15; ++p) {
                    float oc = occ[p];
                    const v8h* ap = (const v8h*)(arow + p * 64 + ob * 16);
                    v8h a0 = ap[0], a1 = ap[1];
#pragma unroll
                    for (int i = 0; i < 8; ++i) {
                        s[i]     += oc * (float)a0[i];
                        s[8 + i] += oc * (float)a1[i];
                    }
                }
#pragma unroll
                for (int i = 0; i < 16; ++i) trow[ob * 16 + i] = (h16)fmaxf(s[i], 0.f);
            }
        }
        __syncthreads();

        // -------- conv2 as im2col GEMM: (y2 x tile) = 8 jobs x 30 K-chunks --------
        for (int job = wave; job < 8; job += 4) {
            int y2 = job >> 1, tn = job & 1;
            v8f acc = fill8(sB2[tn * 16 + (lane & 15)]);
            const h16* aBase = sT1 + y2 * 192;  // 960-contig window, row stride 1536
            for (int kc = 0; kc < 30; ++kc)
                acc = wmma_f16(load_a_frag(aBase + kc * 32, 1536, lane),
                               load_b_frag(W2_p, tn * 30 + kc, lane), acc);
            int n = lane & 15, mh = lane >> 4;
#pragma unroll
            for (int j = 0; j < 8; ++j)  // t2 flat index = o2*4 + y2
                sT2[(j + mh * 8) * 128 + (tn * 16 + n) * 4 + y2] = (h16)fmaxf(acc[j], 0.f);
        }
        __syncthreads();

        // -------- soc_emb = relu(t2 @ W_soc^T + b_soc): 4 tiles x 4 chunks --------
        {
            int tn = wave;
            v8f acc = fill8(sBsoc[tn * 16 + (lane & 15)]);
#pragma unroll
            for (int kc = 0; kc < 4; ++kc)
                acc = wmma_f16(load_a_frag(sT2 + kc * 32, 128, lane),
                               load_b_frag(Wsoc_p, tn * 4 + kc, lane), acc);
            int n = lane & 15, mh = lane >> 4;
#pragma unroll
            for (int j = 0; j < 8; ++j)
                sX[(j + mh * 8) * 128 + 64 + tn * 16 + n] = (h16)fmaxf(acc[j], 0.f);
        }
        __syncthreads();

        // -------- gates = x@W_ih^T + h@W_hh^T + b: 32 tiles x (4+4) chunks --------
        for (int tn = wave; tn < 32; tn += 4) {
            v8f acc = fill8(sBg[tn * 16 + (lane & 15)]);
#pragma unroll
            for (int kc = 0; kc < 4; ++kc)
                acc = wmma_f16(load_a_frag(sX + kc * 32, 128, lane),
                               load_b_frag(Wih_p, tn * 4 + kc, lane), acc);
#pragma unroll
            for (int kc = 0; kc < 4; ++kc)
                acc = wmma_f16(load_a_frag(sH + kc * 32, 128, lane),
                               load_b_frag(Whh_p, tn * 4 + kc, lane), acc);
            int n = lane & 15, mh = lane >> 4;
#pragma unroll
            for (int j = 0; j < 8; ++j)
                sG[(j + mh * 8) * 512 + tn * 16 + n] = acc[j];
        }
        __syncthreads();

        // -------- LSTM elementwise update --------
        for (int i = tid; i < 2048; i += 128) {
            int v = i >> 7, r = i & 127;
            const float* g = sG + v * 512;
            float ig = sigmoidf_(g[r]);
            float fg = sigmoidf_(g[128 + r]);
            float gg = tanhf(g[256 + r]);
            float og = sigmoidf_(g[384 + r]);
            float cn = fg * sC[i] + ig * gg;
            float hn = og * tanhf(cn);
            sC[i] = cn;
            sH[i] = (h16)hn;
        }
        __syncthreads();

        // -------- out = h_new @ W_out^T + b_out (wave 0) --------
        if (wave == 0) {
            v8f acc = fill8(sBout[lane & 15]);
#pragma unroll
            for (int kc = 0; kc < 4; ++kc)
                acc = wmma_f16(load_a_frag(sH + kc * 32, 128, lane),
                               load_b_frag(Wout_p, kc, lane), acc);
            int n = lane & 15, mh = lane >> 4;
            if (n < 5) {
#pragma unroll
                for (int j = 0; j < 8; ++j)
                    out[(t * 256 + v0 + (j + mh * 8)) * 5 + n] = acc[j];
            }
        }
        __syncthreads();
    }
}

// ---------------------------------------------------------------------------
extern "C" void kernel_launch(void* const* d_in, const int* in_sizes, int n_in,
                              void* d_out, int out_size, void* d_ws, size_t ws_size,
                              hipStream_t stream) {
    (void)in_sizes; (void)n_in; (void)out_size; (void)ws_size;
    const float* x_seq   = (const float*)d_in[0];
    const int*   grids   = (const int*)  d_in[1];
    const float* h0      = (const float*)d_in[2];
    const float* c0      = (const float*)d_in[3];
    const float* W_emb   = (const float*)d_in[4];
    const float* b_emb   = (const float*)d_in[5];
    const float* conv1_w = (const float*)d_in[6];
    const float* conv1_b = (const float*)d_in[7];
    const float* conv2_w = (const float*)d_in[8];
    const float* conv2_b = (const float*)d_in[9];
    const float* W_soc   = (const float*)d_in[10];
    const float* b_soc   = (const float*)d_in[11];
    const float* W_ih    = (const float*)d_in[12];
    const float* W_hh    = (const float*)d_in[13];
    const float* b_ih    = (const float*)d_in[14];
    const float* b_hh    = (const float*)d_in[15];
    const float* W_out   = (const float*)d_in[16];
    const float* b_out   = (const float*)d_in[17];
    float* out = (float*)d_out;

    h16* ws = (h16*)d_ws;
    // fragment-packed weight offsets (in halves); each frag = 512 halves
    const size_t OFF_EMB = 0;       // 4 tiles x 1 chunk   =   2048
    const size_t OFF_W1  = 2048;    // 60 x 4              = 122880
    const size_t OFF_W2  = 124928;  // 2 x 30              =  30720
    const size_t OFF_SOC = 155648;  // 4 x 4               =   8192
    const size_t OFF_IH  = 163840;  // 32 x 4              =  65536
    const size_t OFF_HH  = 229376;  // 32 x 4              =  65536
    const size_t OFF_OUT = 294912;  // 1 x 4               =   2048  (end 296960 halves ~ 580KB)

    auto packLaunch = [&](const float* src, size_t off, int mode, int tiles, int chunks) {
        int total = tiles * chunks * 512;
        pack_kernel<<<(total + 255) / 256, 256, 0, stream>>>(src, ws + off, mode, chunks, total);
    };
    packLaunch(W_emb,   OFF_EMB, 0,  4,  1);
    packLaunch(conv1_w, OFF_W1,  1, 60,  4);
    packLaunch(conv2_w, OFF_W2,  2,  2, 30);
    packLaunch(W_soc,   OFF_SOC, 3,  4,  4);
    packLaunch(W_ih,    OFF_IH,  3, 32,  4);
    packLaunch(W_hh,    OFF_HH,  3, 32,  4);
    packLaunch(W_out,   OFF_OUT, 4,  1,  4);

    const int SMEM = 143360;  // ~140KB of the 320KB WGP LDS
    (void)hipFuncSetAttribute((const void*)social_lstm_kernel,
                              hipFuncAttributeMaxDynamicSharedMemorySize, SMEM);
    social_lstm_kernel<<<16, 128, SMEM, stream>>>(
        x_seq, grids, h0, c0,
        b_emb, conv1_b, conv2_b, b_soc, b_ih, b_hh, b_out,
        (const v16h*)(ws + OFF_EMB), (const v16h*)(ws + OFF_W1),
        (const v16h*)(ws + OFF_W2),  (const v16h*)(ws + OFF_SOC),
        (const v16h*)(ws + OFF_IH),  (const v16h*)(ws + OFF_HH),
        (const v16h*)(ws + OFF_OUT), out);
}
